// gnn_rnn_diagnoser_38637525795084
// MI455X (gfx1250) — compile-verified, matrix operand order checked by reading, not run on hardware
//
#include <hip/hip_runtime.h>
#include <hip/hip_bf16.h>

typedef __attribute__((ext_vector_type(16))) __bf16 v16bf;
typedef __attribute__((ext_vector_type(8)))  __bf16 v8bf;
typedef __attribute__((ext_vector_type(8)))  float  v8f;

#define B_TOTAL 1024
#define T_STEPS 128
#define N_NODES 5
#define FML 32
#define IN_F 165
#define KP1 192          // padded input feature K (6 * 32)
#define KC1 6            // K-chunks for 192
#define KC2 8            // K-chunks for 256
#define BM 16            // batch rows per workgroup
#define NWG (B_TOTAL / BM)

// workspace byte offsets
#define OFF_WF1 0u
#define OFF_WF2 (OFF_WF1 + 5u*KP1*256u*2u)          // 491520
#define OFF_WF3 (OFF_WF2 + 5u*256u*256u*2u)         // 1146880
#define OFF_WO1 (OFF_WF3 + 5u*256u*32u*2u)          // 1228800
#define OFF_H   (OFF_WO1 + (unsigned)KP1*512u*2u)   // 1425408
#define OFF_SUM (OFF_H   + 1024u*512u*4u)           // 3522560
#define OFF_SSQ (OFF_SUM + 512u*4u)                 // 3524608

// ---------------------------------------------------------------------------
// Prep: f32 row-major [mat][K][N] -> bf16 WMMA-B fragment layout
// dst flat order: [mat][nt][kc][512], within-frag: lane*16 + r*2 + h
// kr(lane,r,h) = ((r&4)<<2) + ((r&3)<<1) + h + ((lane>>4)<<3)
// ---------------------------------------------------------------------------
__global__ __launch_bounds__(256) void prep_frag(const float* __restrict__ src,
                                                 __bf16* __restrict__ dst,
                                                 int Ksrc, int Kpad, int N,
                                                 int srcStride, long total) {
  long id = (long)blockIdx.x * blockDim.x + threadIdx.x;
  if (id >= total) return;
  int KC = Kpad >> 5, NT = N >> 4;
  int within = (int)(id & 511);
  long f = id >> 9;
  int kc = (int)(f % KC); f /= KC;
  int nt = (int)(f % NT);
  int mat = (int)(f / NT);
  int lane = within >> 4, idx = within & 15;
  int r = idx >> 1, h = idx & 1;
  int kr = ((r & 4) << 2) + ((r & 3) << 1) + h + ((lane >> 4) << 3);
  int k = (kc << 5) + kr;
  int n = (nt << 4) + (lane & 15);
  float val = (k < Ksrc) ? src[(long)mat * srcStride + (long)k * N + n] : 0.f;
  dst[id] = (__bf16)val;
}

__global__ void zero_f32(float* p, int n) {
  int i = blockIdx.x * blockDim.x + threadIdx.x;
  if (i < n) p[i] = 0.f;
}

// A fragment (16x32 bf16) from LDS, row-major [16][stride]:
// lane l: M = l&15, K base = (l>>4)*8 ; regs 0-3 = K..K+7, regs 4-7 = K+16..K+23
__device__ __forceinline__ v16bf load_a_frag(const __bf16* lds, int lane,
                                             int stride, int kc) {
  int m = lane & 15;
  int kb = (kc << 5) + ((lane >> 4) << 3);
  union { v16bf v; v8bf h[2]; } u;
  u.h[0] = *(const v8bf*)(lds + m * stride + kb);
  u.h[1] = *(const v8bf*)(lds + m * stride + kb + 16);
  return u.v;
}

__device__ __forceinline__ v16bf load_b_frag(const __bf16* g, int lane) {
  return *(const v16bf*)(g + lane * 16);
}

#define WMMA_BF16(acc, a, b) \
  acc = __builtin_amdgcn_wmma_f32_16x16x32_bf16(false, a, false, b, (short)0, acc, false, false)

// ---------------------------------------------------------------------------
// Recurrence: 64 WGs x 128 threads (4 waves). 16 batch rows per WG.
// LDS: inp[16][192] (x_t | state | pad), h1[16][256], h2[16][256]
// ---------------------------------------------------------------------------
__global__ __launch_bounds__(128) void recur_kernel(
    const float* __restrict__ x,
    const float* __restrict__ bf1, const float* __restrict__ bf2,
    const float* __restrict__ bf3, const float* __restrict__ bo1,
    const __bf16* __restrict__ wf1, const __bf16* __restrict__ wf2,
    const __bf16* __restrict__ wf3, const __bf16* __restrict__ wo1,
    float* __restrict__ hbuf, float* __restrict__ gsum, float* __restrict__ gssq) {
  __shared__ __bf16 inp[BM * KP1];
  __shared__ __bf16 h1[BM * 256];
  __shared__ __bf16 h2[BM * 256];

  const int tid = threadIdx.x;
  const int lane = tid & 31;
  const int wave = tid >> 5;
  const int b0 = blockIdx.x * BM;
  const int nl = lane & 15;       // N within tile
  const int mo = (lane >> 4) << 3; // M offset for C/D regs

  // init state + pad to zero
  for (int i = tid; i < BM * KP1; i += 128) {
    if ((i % KP1) >= N_NODES) inp[i] = (__bf16)0.f;
  }
  __syncthreads();

  for (int t = 0; t < T_STEPS; ++t) {
    // x_t -> inp[:, 0:5]
    if (tid < BM * N_NODES) {
      int m = tid / N_NODES, c = tid % N_NODES;
      inp[m * KP1 + c] = (__bf16)x[((b0 + m) * N_NODES + c) * T_STEPS + t];
    }
    __syncthreads();

    for (int node = 0; node < N_NODES; ++node) {
      // ---- GEMM1: inp[16x192] x Wf1[node][192x256] -> h1 (relu) ----
      {
        const __bf16* wbase = wf1 + (size_t)node * (KP1 * 256);
        if (lane == 0)
          __builtin_prefetch((const void*)(wf2 + (size_t)node * (256 * 256)), 0, 1);
        v16bf a[KC1];
#pragma unroll
        for (int kc = 0; kc < KC1; ++kc) a[kc] = load_a_frag(inp, lane, KP1, kc);
#pragma unroll
        for (int nt0 = 0; nt0 < 4; ++nt0) {
          int nt = wave * 4 + nt0;
          v8f acc = {};
          const __bf16* wp = wbase + nt * (KC1 * 512);
#pragma unroll
          for (int kc = 0; kc < KC1; ++kc) {
            v16bf b = load_b_frag(wp, lane); wp += 512;
            WMMA_BF16(acc, a[kc], b);
          }
          int n = nt * 16 + nl;
          float bias = bf1[node * 256 + n];
#pragma unroll
          for (int v = 0; v < 8; ++v) {
            float val = fmaxf(acc[v] + bias, 0.f);
            h1[(mo + v) * 256 + n] = (__bf16)val;
          }
        }
      }
      __syncthreads();

      // ---- GEMM2: h1[16x256] x Wf2[node][256x256] -> h2 (relu) ----
      {
        const __bf16* wbase = wf2 + (size_t)node * (256 * 256);
        if (lane == 0)
          __builtin_prefetch((const void*)(wf3 + (size_t)node * (256 * 32)), 0, 1);
        v16bf a[KC2];
#pragma unroll
        for (int kc = 0; kc < KC2; ++kc) a[kc] = load_a_frag(h1, lane, 256, kc);
#pragma unroll
        for (int nt0 = 0; nt0 < 4; ++nt0) {
          int nt = wave * 4 + nt0;
          v8f acc = {};
          const __bf16* wp = wbase + nt * (KC2 * 512);
#pragma unroll
          for (int kc = 0; kc < KC2; ++kc) {
            v16bf b = load_b_frag(wp, lane); wp += 512;
            WMMA_BF16(acc, a[kc], b);
          }
          int n = nt * 16 + nl;
          float bias = bf2[node * 256 + n];
#pragma unroll
          for (int v = 0; v < 8; ++v) {
            float val = fmaxf(acc[v] + bias, 0.f);
            h2[(mo + v) * 256 + n] = (__bf16)val;
          }
        }
      }
      __syncthreads();

      // ---- GEMM3: h2[16x256] x Wf3[node][256x32] -> state (relu) ----
      if (wave < 2) {
        v16bf a[KC2];
#pragma unroll
        for (int kc = 0; kc < KC2; ++kc) a[kc] = load_a_frag(h2, lane, 256, kc);
        int nt = wave;
        v8f acc = {};
        const __bf16* wp = wf3 + (size_t)node * (256 * 32) + nt * (KC2 * 512);
#pragma unroll
        for (int kc = 0; kc < KC2; ++kc) {
          v16bf b = load_b_frag(wp, lane); wp += 512;
          WMMA_BF16(acc, a[kc], b);
        }
        int n = nt * 16 + nl;
        float bias = bf3[node * FML + n];
        int col = N_NODES + node * FML + n; // Gauss-Seidel in-place state update
#pragma unroll
        for (int v = 0; v < 8; ++v) {
          float val = fmaxf(acc[v] + bias, 0.f);
          inp[(mo + v) * KP1 + col] = (__bf16)val;
        }
      }
      __syncthreads();
    }
  }

  // ---- Head GEMM: feat[16x192] x Wo1[192x512] -> hbuf (relu) + BN stats ----
  {
    v16bf a[KC1];
#pragma unroll
    for (int kc = 0; kc < KC1; ++kc) a[kc] = load_a_frag(inp, lane, KP1, kc);
#pragma unroll
    for (int nt0 = 0; nt0 < 8; ++nt0) {
      int nt = wave * 8 + nt0;
      v8f acc = {};
      const __bf16* wp = wo1 + nt * (KC1 * 512);
#pragma unroll
      for (int kc = 0; kc < KC1; ++kc) {
        v16bf b = load_b_frag(wp, lane); wp += 512;
        WMMA_BF16(acc, a[kc], b);
      }
      int n = nt * 16 + nl;
      float bias = bo1[n];
      float s = 0.f, s2 = 0.f;
#pragma unroll
      for (int v = 0; v < 8; ++v) {
        float val = fmaxf(acc[v] + bias, 0.f);
        hbuf[(size_t)(b0 + mo + v) * 512 + n] = val;
        s += val; s2 += val * val;
      }
      s  += __shfl_xor(s, 16, 32);   // combine M=0..7 with M=8..15 halves
      s2 += __shfl_xor(s2, 16, 32);
      if (lane < 16) {
        atomicAdd(&gsum[n], s);
        atomicAdd(&gssq[n], s2);
      }
    }
  }
}

// ---------------------------------------------------------------------------
// Head finish: BN (training stats) -> logits -> softmax
// ---------------------------------------------------------------------------
__global__ __launch_bounds__(128) void head_finish(
    const float* __restrict__ hbuf, const float* __restrict__ gsum,
    const float* __restrict__ gssq, const float* __restrict__ gamma,
    const float* __restrict__ beta, const float* __restrict__ Wo2,
    const float* __restrict__ bo2, float* __restrict__ out) {
  int b = blockIdx.x * blockDim.x + threadIdx.x;
  if (b >= B_TOTAL) return;
  float logits[7];
#pragma unroll
  for (int j = 0; j < 7; ++j) logits[j] = bo2[j];
  for (int c = 0; c < 512; ++c) {
    float mu = gsum[c] * (1.f / B_TOTAL);
    float var = gssq[c] * (1.f / B_TOTAL) - mu * mu;
    float inv = rsqrtf(var + 1e-5f);
    float xn = (hbuf[(size_t)b * 512 + c] - mu) * inv * gamma[c] + beta[c];
#pragma unroll
    for (int j = 0; j < 7; ++j) logits[j] += xn * Wo2[c * 7 + j];
  }
  float mx = logits[0];
#pragma unroll
  for (int j = 1; j < 7; ++j) mx = fmaxf(mx, logits[j]);
  float e[7], s = 0.f;
#pragma unroll
  for (int j = 0; j < 7; ++j) { e[j] = __expf(logits[j] - mx); s += e[j]; }
  float inv = 1.f / s;
#pragma unroll
  for (int j = 0; j < 7; ++j) out[b * 7 + j] = e[j] * inv;
}

extern "C" void kernel_launch(void* const* d_in, const int* in_sizes, int n_in,
                              void* d_out, int out_size, void* d_ws, size_t ws_size,
                              hipStream_t stream) {
  const float* x   = (const float*)d_in[0];
  const float* Wf1 = (const float*)d_in[1];
  const float* bf1 = (const float*)d_in[2];
  const float* Wf2 = (const float*)d_in[3];
  const float* bf2 = (const float*)d_in[4];
  const float* Wf3 = (const float*)d_in[5];
  const float* bf3 = (const float*)d_in[6];
  const float* Wo1 = (const float*)d_in[7];
  const float* bo1 = (const float*)d_in[8];
  const float* gamma = (const float*)d_in[9];
  const float* beta  = (const float*)d_in[10];
  const float* Wo2 = (const float*)d_in[11];
  const float* bo2 = (const float*)d_in[12];

  char* ws = (char*)d_ws;
  __bf16* wf1f = (__bf16*)(ws + OFF_WF1);
  __bf16* wf2f = (__bf16*)(ws + OFF_WF2);
  __bf16* wf3f = (__bf16*)(ws + OFF_WF3);
  __bf16* wo1f = (__bf16*)(ws + OFF_WO1);
  float* hbuf  = (float*)(ws + OFF_H);
  float* gsum  = (float*)(ws + OFF_SUM);
  float* gssq  = (float*)(ws + OFF_SSQ);

  // weight prep (f32 -> bf16 WMMA fragment layout, L2-resident afterwards)
  prep_frag<<<(5L*KP1*256 + 255) / 256, 256, 0, stream>>>(Wf1, wf1f, IN_F, KP1, 256, IN_F * 256, 5L * KP1 * 256);
  prep_frag<<<(5L*256*256 + 255) / 256, 256, 0, stream>>>(Wf2, wf2f, 256, 256, 256, 256 * 256, 5L * 256 * 256);
  prep_frag<<<(5L*256*32 + 255) / 256, 256, 0, stream>>>(Wf3, wf3f, 256, 256, 32, 256 * 32, 5L * 256 * 32);
  prep_frag<<<((long)KP1*512 + 255) / 256, 256, 0, stream>>>(Wo1, wo1f, IN_F, KP1, 512, IN_F * 512, (long)KP1 * 512);
  zero_f32<<<4, 256, 0, stream>>>(gsum, 1024); // gsum + gssq are contiguous

  recur_kernel<<<NWG, 128, 0, stream>>>(x, bf1, bf2, bf3, bo1,
                                        wf1f, wf2f, wf3f, wo1f,
                                        hbuf, gsum, gssq);
  head_finish<<<(B_TOTAL + 127) / 128, 128, 0, stream>>>(hbuf, gsum, gssq, gamma,
                                                         beta, Wo2, bo2, (float*)d_out);
}
